// CrossAttentionFuser_65798898975031
// MI455X (gfx1250) — matrix operand order repaired and checked by reading, not compiled, not run on hardware
//
#include <hip/hip_runtime.h>
#include <hip/hip_bf16.h>

// ---------------------------------------------------------------------------
// CrossAttentionFuser for MI455X (gfx1250): all GEMMs + attention via
// v_wmma_f32_16x16x32_bf16. Wave32; 32x32 output block per wave (4 WMMA /
// k-step with 2x operand reuse each way); native f32<->bf16 cvt.
// ---------------------------------------------------------------------------

typedef __attribute__((ext_vector_type(16))) __bf16 v16bf;
typedef __attribute__((ext_vector_type(8)))  __bf16 v8bf;
typedef __attribute__((ext_vector_type(8)))  float  v8f;
typedef __attribute__((ext_vector_type(4)))  float  v4f;

#define BB     4
#define SS     4096
#define S2S    2048
#define DD     1024
#define SCAFD  768
#define HH     8
#define HDIM   128
#define KPAD   2560        // K (2549) padded to multiple of 32
#define QT16   160         // KPAD / 16  (attention tiles)
#define QT32   80          // KPAD / 32  (GEMM tiles)

static __device__ __forceinline__ __bf16 f2bf(float f) { return (__bf16)f; }
static __device__ __forceinline__ float  bf2f(__bf16 b) { return (float)b; }

static __device__ __forceinline__ v8f wmma_bf16(v16bf a, v16bf b, v8f c) {
  // (neg_a, A, neg_b, B, c_mod, C, reuse_a, reuse_b)
  return __builtin_amdgcn_wmma_f32_16x16x32_bf16(false, a, false, b, (short)0, c, false, false);
}

// A operand (16x32) from fp32 row-major activation row. Lane holds row l%16.
// halves 0..15 -> k = klo..klo+15 (contiguous), klo = hi?16:0.
static __device__ __forceinline__ v16bf load_a_row_f32(const float* __restrict__ rowp,
                                                       int k0, bool hi) {
  const int klo = k0 + (hi ? 16 : 0);
  const v4f* p0 = (const v4f*)(rowp + klo);
  const v4f* p1 = (const v4f*)(rowp + klo + 8);
  v4f x0 = p0[0], x1 = p0[1], y0 = p1[0], y1 = p1[1];
  v16bf a;
#pragma unroll
  for (int i = 0; i < 4; ++i) {
    a[i]      = f2bf(x0[i]);
    a[4 + i]  = f2bf(x1[i]);
    a[8 + i]  = f2bf(y0[i]);
    a[12 + i] = f2bf(y1[i]);
  }
  return a;
}

// Same A layout but from a bf16 row.
static __device__ __forceinline__ v16bf load_a_row_bf16(const __bf16* __restrict__ rowp,
                                                        int k0, bool hi) {
  const int klo = k0 + (hi ? 16 : 0);
  v8bf lo = *(const v8bf*)(rowp + klo);
  v8bf h8 = *(const v8bf*)(rowp + klo + 8);
  v16bf a;
#pragma unroll
  for (int i = 0; i < 8; ++i) { a[i] = lo[i]; a[8 + i] = h8[i]; }
  return a;
}

// B operand (32x16) from fp32 weight stored (N,K) row-major.
// Lane holds column n = n0 + l%16 ; halves h -> k = k0 + (hi?16:0) + h (contiguous).
static __device__ __forceinline__ v16bf load_b_f32(const float* __restrict__ W, int ldk,
                                                   int n, int k0, bool hi) {
  const float* p = W + (long)n * ldk + k0 + (hi ? 16 : 0);
  const v4f* q = (const v4f*)p;
  v16bf b;
#pragma unroll
  for (int j = 0; j < 4; ++j) {
    v4f x = q[j];
#pragma unroll
    for (int i = 0; i < 4; ++i) b[j * 4 + i] = f2bf(x[i]);
  }
  return b;
}

// ---------------------------------------------------------------------------
__global__ void k_init(float* conf) {
  if (threadIdx.x == 0 && blockIdx.x == 0) conf[0] = 0.f;
}

// scaf[m][n] = scaffold_hidden[m,:] . scaffold_proj_w[n,:] + b[n]  (8192x1024x768)
__global__ __launch_bounds__(32)
void k_scaf_proj(const float* __restrict__ X, const float* __restrict__ W,
                 const float* __restrict__ bias, __bf16* __restrict__ scaf) {
  const int lane = threadIdx.x;
  const bool hi = lane >= 16;
  const int ln = lane & 15;
  const int m0 = blockIdx.x * 32;
  const int n0 = blockIdx.y * 32;
  const float* rp0 = X + (long)(m0 + ln) * SCAFD;
  const float* rp1 = X + (long)(m0 + 16 + ln) * SCAFD;
  const int na = n0 + ln, nb = n0 + 16 + ln;
  v8f c00 = {}, c01 = {}, c10 = {}, c11 = {};
  for (int k0 = 0; k0 < SCAFD; k0 += 32) {
    __builtin_prefetch(rp0 + k0 + 128, 0, 0);
    __builtin_prefetch(rp1 + k0 + 128, 0, 0);
    v16bf a0 = load_a_row_f32(rp0, k0, hi);
    v16bf a1 = load_a_row_f32(rp1, k0, hi);
    v16bf b0 = load_b_f32(W, SCAFD, na, k0, hi);
    v16bf b1 = load_b_f32(W, SCAFD, nb, k0, hi);
    c00 = wmma_bf16(a0, b0, c00);
    c01 = wmma_bf16(a0, b1, c01);
    c10 = wmma_bf16(a1, b0, c10);
    c11 = wmma_bf16(a1, b1, c11);
  }
  const float ba = bias[na], bbn = bias[nb];
#pragma unroll
  for (int v = 0; v < 8; ++v) {
    const int r0 = m0 + v + (hi ? 8 : 0);
    const int r1 = r0 + 16;
    scaf[(long)r0 * DD + na] = f2bf(c00[v] + ba);
    scaf[(long)r0 * DD + nb] = f2bf(c01[v] + bbn);
    scaf[(long)r1 * DD + na] = f2bf(c10[v] + ba);
    scaf[(long)r1 * DD + nb] = f2bf(c11[v] + bbn);
  }
}

// mean row norm accumulation
__global__ __launch_bounds__(256)
void k_conf(const __bf16* __restrict__ scaf, float* conf) {
  __shared__ float red[256];
  const __bf16* p = scaf + (long)blockIdx.x * DD;
  float ss = 0.f;
  for (int i = threadIdx.x; i < DD; i += 256) { float x = bf2f(p[i]); ss += x * x; }
  red[threadIdx.x] = ss;
  __syncthreads();
  for (int off = 128; off > 0; off >>= 1) {
    if (threadIdx.x < off) red[threadIdx.x] += red[threadIdx.x + off];
    __syncthreads();
  }
  if (threadIdx.x == 0) atomicAdd(conf, sqrtf(red[0]));
}

// per-token topk score + gate (one wave per token row)
__global__ __launch_bounds__(256)
void k_scores(const float* __restrict__ base, const float* __restrict__ tw,
              const float* __restrict__ tb, const float* __restrict__ gw,
              const float* __restrict__ gb, float* __restrict__ scores,
              float* __restrict__ gates) {
  const int wid = (int)((blockIdx.x * blockDim.x + threadIdx.x) >> 5);
  const int lane = threadIdx.x & 31;
  if (wid >= BB * SS) return;
  const float* p = base + (long)wid * DD;
  float s = 0.f, g = 0.f;
  for (int i = lane; i < DD; i += 32) { float x = p[i]; s += x * tw[i]; g += x * gw[i]; }
#pragma unroll
  for (int m = 16; m > 0; m >>= 1) { s += __shfl_xor(s, m, 32); g += __shfl_xor(g, m, 32); }
  if (lane == 0) {
    scores[wid] = s + tb[0];
    gates[wid]  = 1.f / (1.f + __expf(-(g + gb[0])));
  }
}

// per-batch descending bitonic sort of 4096 (score,index), keep top-K index set
__global__ __launch_bounds__(1024)
void k_topk(const float* __restrict__ scores, int* __restrict__ idxb, int K) {
  __shared__ float sv[SS];
  __shared__ int   si[SS];
  const int b = blockIdx.x;
  const float* sc = scores + (long)b * SS;
  for (int i = threadIdx.x; i < SS; i += 1024) { sv[i] = sc[i]; si[i] = i; }
  __syncthreads();
  for (int kk = 2; kk <= SS; kk <<= 1) {
    for (int j = kk >> 1; j > 0; j >>= 1) {
      for (int t = threadIdx.x; t < SS / 2; t += 1024) {
        const int i = ((t & ~(j - 1)) << 1) | (t & (j - 1));
        const int ixj = i | j;
        const bool up = ((i & kk) == 0);       // up => larger first (descending)
        float a = sv[i], c = sv[ixj];
        int ia = si[i], ic = si[ixj];
        const bool agtb = (a > c) || (a == c && ia < ic);
        if (up ? !agtb : agtb) { sv[i] = c; sv[ixj] = a; si[i] = ic; si[ixj] = ia; }
      }
      __syncthreads();
    }
  }
  for (int j = threadIdx.x; j < KPAD; j += 1024)
    idxb[b * KPAD + j] = (j < K) ? si[j] : 0;
}

// q[b,j,:] = base[b, idx[b,j], :] . Wq^T + bq   (gathered-A GEMM, 32x32 blocks)
__global__ __launch_bounds__(32)
void k_qproj(const float* __restrict__ base, const int* __restrict__ idxb,
             const float* __restrict__ W, const float* __restrict__ bias,
             __bf16* __restrict__ qb) {
  const int lane = threadIdx.x;
  const bool hi = lane >= 16;
  const int ln = lane & 15;
  const int b = blockIdx.x / QT32;
  const int m0 = (blockIdx.x % QT32) * 32;
  const int n0 = blockIdx.y * 32;
  const int sr0 = idxb[b * KPAD + m0 + ln];
  const int sr1 = idxb[b * KPAD + m0 + 16 + ln];
  const float* rp0 = base + ((long)b * SS + sr0) * DD;
  const float* rp1 = base + ((long)b * SS + sr1) * DD;
  const int na = n0 + ln, nb = n0 + 16 + ln;
  v8f c00 = {}, c01 = {}, c10 = {}, c11 = {};
  for (int k0 = 0; k0 < DD; k0 += 32) {
    __builtin_prefetch(rp0 + k0 + 128, 0, 0);
    __builtin_prefetch(rp1 + k0 + 128, 0, 0);
    v16bf a0 = load_a_row_f32(rp0, k0, hi);
    v16bf a1 = load_a_row_f32(rp1, k0, hi);
    v16bf b0 = load_b_f32(W, DD, na, k0, hi);
    v16bf b1 = load_b_f32(W, DD, nb, k0, hi);
    c00 = wmma_bf16(a0, b0, c00);
    c01 = wmma_bf16(a0, b1, c01);
    c10 = wmma_bf16(a1, b0, c10);
    c11 = wmma_bf16(a1, b1, c11);
  }
  const float ba = bias[na], bbn = bias[nb];
#pragma unroll
  for (int v = 0; v < 8; ++v) {
    const int j0 = m0 + v + (hi ? 8 : 0);
    const int j1 = j0 + 16;
    qb[((long)b * KPAD + j0) * DD + na] = f2bf(c00[v] + ba);
    qb[((long)b * KPAD + j0) * DD + nb] = f2bf(c01[v] + bbn);
    qb[((long)b * KPAD + j1) * DD + na] = f2bf(c10[v] + ba);
    qb[((long)b * KPAD + j1) * DD + nb] = f2bf(c11[v] + bbn);
  }
}

// K,V projection from scaf. N = 2048 over [Wk;Wv]; V stored transposed [b][d][s2].
__global__ __launch_bounds__(32)
void k_kvproj(const __bf16* __restrict__ scaf, const float* __restrict__ W,
              const float* __restrict__ bias, __bf16* __restrict__ kb,
              __bf16* __restrict__ vT) {
  const int lane = threadIdx.x;
  const bool hi = lane >= 16;
  const int ln = lane & 15;
  const int m0 = blockIdx.x * 32;       // 0..8191 global (b,s2) row
  const int n0 = blockIdx.y * 32;       // 0..2047 (tile is entirely K or entirely V)
  const __bf16* rp0 = scaf + (long)(m0 + ln) * DD;
  const __bf16* rp1 = scaf + (long)(m0 + 16 + ln) * DD;
  const int na = n0 + ln, nb = n0 + 16 + ln;
  v8f c00 = {}, c01 = {}, c10 = {}, c11 = {};
  for (int k0 = 0; k0 < DD; k0 += 32) {
    __builtin_prefetch(rp0 + k0 + 256, 0, 0);
    __builtin_prefetch(rp1 + k0 + 256, 0, 0);
    v16bf a0 = load_a_row_bf16(rp0, k0, hi);
    v16bf a1 = load_a_row_bf16(rp1, k0, hi);
    v16bf b0 = load_b_f32(W, DD, na, k0, hi);
    v16bf b1 = load_b_f32(W, DD, nb, k0, hi);
    c00 = wmma_bf16(a0, b0, c00);
    c01 = wmma_bf16(a0, b1, c01);
    c10 = wmma_bf16(a1, b0, c10);
    c11 = wmma_bf16(a1, b1, c11);
  }
  const float ba = bias[na], bbn = bias[nb];
  if (n0 < DD) {                         // K path: row-major [b*s2][d]
#pragma unroll
    for (int v = 0; v < 8; ++v) {
      const int r0 = m0 + v + (hi ? 8 : 0);
      const int r1 = r0 + 16;
      kb[(long)r0 * DD + na] = f2bf(c00[v] + ba);
      kb[(long)r0 * DD + nb] = f2bf(c01[v] + bbn);
      kb[(long)r1 * DD + na] = f2bf(c10[v] + ba);
      kb[(long)r1 * DD + nb] = f2bf(c11[v] + bbn);
    }
  } else {                               // V path: transposed [b][d][s2]
    const int b = m0 / S2S;
    const int s20 = m0 % S2S;
    const int da = na - DD, db = nb - DD;
#pragma unroll
    for (int v = 0; v < 8; ++v) {
      const int s0 = s20 + v + (hi ? 8 : 0);
      const int s1 = s0 + 16;
      vT[((long)b * DD + da) * S2S + s0] = f2bf(c00[v] + ba);
      vT[((long)b * DD + db) * S2S + s0] = f2bf(c01[v] + bbn);
      vT[((long)b * DD + da) * S2S + s1] = f2bf(c10[v] + ba);
      vT[((long)b * DD + db) * S2S + s1] = f2bf(c11[v] + bbn);
    }
  }
}

// flash attention: one wave per (b, h, 16-query tile); stream 32 keys per step
__global__ __launch_bounds__(32)
void k_attn(const __bf16* __restrict__ qb, const __bf16* __restrict__ kb,
            const __bf16* __restrict__ vT, __bf16* __restrict__ ob) {
  __shared__ __align__(32) __bf16 P[16 * 32];
  const int lane = threadIdx.x;
  const bool hi = lane >= 16;
  const int ln = lane & 15;
  int blk = blockIdx.x;
  const int qt = blk % QT16; blk /= QT16;
  const int h = blk % HH;
  const int b = blk / HH;
  const int m0 = qt * 16;

  const __bf16* qrow = qb + ((long)b * KPAD + (m0 + ln)) * DD + h * HDIM;
  v16bf qa[4];
#pragma unroll
  for (int ks = 0; ks < 4; ++ks) qa[ks] = load_a_row_bf16(qrow, ks * 32, hi);

  float M[8], L[8];
  v8f acc[8];
#pragma unroll
  for (int v = 0; v < 8; ++v) { M[v] = -1e30f; L[v] = 0.f; }
#pragma unroll
  for (int t = 0; t < 8; ++t) acc[t] = (v8f){};

  const float scale = 0.088388347648318447f;  // 1/sqrt(128)
  const __bf16* kbase = kb + (long)b * S2S * DD + h * HDIM;
  const __bf16* vbase = vT + ((long)b * DD + h * HDIM) * S2S;

  for (int key0 = 0; key0 < S2S; key0 += 32) {
    v8f s0 = {}, s1 = {};
#pragma unroll
    for (int ks = 0; ks < 4; ++ks) {
      v16bf bk = *(const v16bf*)(kbase + (long)(key0 + ln) * DD + ks * 32 + (hi ? 16 : 0));
      s0 = wmma_bf16(qa[ks], bk, s0);
    }
#pragma unroll
    for (int ks = 0; ks < 4; ++ks) {
      v16bf bk = *(const v16bf*)(kbase + (long)(key0 + 16 + ln) * DD + ks * 32 + (hi ? 16 : 0));
      s1 = wmma_bf16(qa[ks], bk, s1);
    }
    // online softmax per row (row v (+8 for hi lanes) lives across one 16-lane half)
#pragma unroll
    for (int v = 0; v < 8; ++v) {
      float x0 = s0[v] * scale, x1 = s1[v] * scale;
      float m = fmaxf(x0, x1);
#pragma unroll
      for (int mm = 1; mm < 16; mm <<= 1) m = fmaxf(m, __shfl_xor(m, mm, 32));
      const float newM = fmaxf(M[v], m);
      const float f = __expf(M[v] - newM);
      M[v] = newM;
      const float p0 = __expf(x0 - newM), p1 = __expf(x1 - newM);
      s0[v] = p0; s1[v] = p1;
      float ps = p0 + p1;
#pragma unroll
      for (int mm = 1; mm < 16; mm <<= 1) ps += __shfl_xor(ps, mm, 32);
      L[v] = L[v] * f + ps;
#pragma unroll
      for (int t = 0; t < 8; ++t) acc[t][v] *= f;
    }
    // C-layout P -> LDS -> A-layout
#pragma unroll
    for (int v = 0; v < 8; ++v) {
      const int r = v + (hi ? 8 : 0);
      P[r * 32 + ln]      = f2bf(s0[v]);
      P[r * 32 + 16 + ln] = f2bf(s1[v]);
    }
    __syncthreads();
    v16bf pa = load_a_row_bf16(P + ln * 32, 0, hi);
#pragma unroll
    for (int t = 0; t < 8; ++t) {
      v16bf vb = *(const v16bf*)(vbase + (long)(t * 16 + ln) * S2S + key0 + (hi ? 16 : 0));
      acc[t] = wmma_bf16(pa, vb, acc[t]);
    }
    __syncthreads();
  }
#pragma unroll
  for (int v = 0; v < 8; ++v) {
    const int j = m0 + v + (hi ? 8 : 0);
    const float invl = 1.f / L[v];
#pragma unroll
    for (int t = 0; t < 8; ++t)
      ob[((long)b * KPAD + j) * DD + h * HDIM + t * 16 + ln] = f2bf(acc[t][v] * invl);
  }
}

// default fill: out = fuse ? base*(1 + 0.5*gate) : base   (attn_full==base off-topk)
__global__ __launch_bounds__(256)
void k_fill(const float* __restrict__ base, const float* __restrict__ gates,
            const float* __restrict__ confacc, const float* __restrict__ thr,
            float* __restrict__ out) {
  const long i = (long)blockIdx.x * 256 + threadIdx.x;
  if (i >= (long)BB * SS * DD) return;
  const float bv = base[i];
  const bool fuse = (confacc[0] * (1.f / (BB * S2S))) > thr[0];
  const float g = gates[i >> 10];
  out[i] = fuse ? bv * (1.f + 0.5f * g) : bv;
}

// out-proj GEMM for the selected rows + gated scatter-blend into d_out (32x32)
__global__ __launch_bounds__(32)
void k_oproj_fuse(const __bf16* __restrict__ ob, const float* __restrict__ W,
                  const float* __restrict__ bias, const int* __restrict__ idxb,
                  const float* __restrict__ gates, const float* __restrict__ base,
                  const float* __restrict__ confacc, const float* __restrict__ thr,
                  float* __restrict__ out, int K) {
  const int lane = threadIdx.x;
  const bool hi = lane >= 16;
  const int ln = lane & 15;
  const int b = blockIdx.x / QT32;
  const int m0 = (blockIdx.x % QT32) * 32;
  const int n0 = blockIdx.y * 32;
  const __bf16* rp0 = ob + ((long)b * KPAD + m0 + ln) * DD;
  const __bf16* rp1 = ob + ((long)b * KPAD + m0 + 16 + ln) * DD;
  const int na = n0 + ln, nb = n0 + 16 + ln;
  v8f c00 = {}, c01 = {}, c10 = {}, c11 = {};
  for (int k0 = 0; k0 < DD; k0 += 32) {
    __builtin_prefetch(rp0 + k0 + 256, 0, 0);
    __builtin_prefetch(rp1 + k0 + 256, 0, 0);
    v16bf a0 = load_a_row_bf16(rp0, k0, hi);
    v16bf a1 = load_a_row_bf16(rp1, k0, hi);
    v16bf b0 = load_b_f32(W, DD, na, k0, hi);
    v16bf b1 = load_b_f32(W, DD, nb, k0, hi);
    c00 = wmma_bf16(a0, b0, c00);
    c01 = wmma_bf16(a0, b1, c01);
    c10 = wmma_bf16(a1, b0, c10);
    c11 = wmma_bf16(a1, b1, c11);
  }
  const float ba = bias[na], bbn = bias[nb];
  const bool fuse = (confacc[0] * (1.f / (BB * S2S))) > thr[0];
#pragma unroll
  for (int v = 0; v < 8; ++v) {
    const int j0 = m0 + v + (hi ? 8 : 0);
    const int j1 = j0 + 16;
    if (j0 < K) {
      const int s = idxb[b * KPAD + j0];
      const float g = gates[b * SS + s];
      long o0 = ((long)b * SS + s) * DD + na;
      long o1 = ((long)b * SS + s) * DD + nb;
      out[o0] = fuse ? (base[o0] + 0.5f * g * (c00[v] + ba)) : base[o0];
      out[o1] = fuse ? (base[o1] + 0.5f * g * (c01[v] + bbn)) : base[o1];
    }
    if (j1 < K) {
      const int s = idxb[b * KPAD + j1];
      const float g = gates[b * SS + s];
      long o0 = ((long)b * SS + s) * DD + na;
      long o1 = ((long)b * SS + s) * DD + nb;
      out[o0] = fuse ? (base[o0] + 0.5f * g * (c10[v] + ba)) : base[o0];
      out[o1] = fuse ? (base[o1] + 0.5f * g * (c11[v] + bbn)) : base[o1];
    }
  }
}

// ---------------------------------------------------------------------------
extern "C" void kernel_launch(void* const* d_in, const int* in_sizes, int n_in,
                              void* d_out, int out_size, void* d_ws, size_t ws_size,
                              hipStream_t stream) {
  (void)in_sizes; (void)n_in; (void)out_size; (void)ws_size;
  const float* base = (const float*)d_in[0];
  const float* scafh = (const float*)d_in[1];
  const float* spw = (const float*)d_in[2];
  const float* spb = (const float*)d_in[3];
  const float* tw  = (const float*)d_in[4];
  const float* tb  = (const float*)d_in[5];
  const float* ipw = (const float*)d_in[6];
  const float* ipb = (const float*)d_in[7];
  const float* opw = (const float*)d_in[8];
  const float* opb = (const float*)d_in[9];
  const float* gw  = (const float*)d_in[10];
  const float* gb  = (const float*)d_in[11];
  const float* thr = (const float*)d_in[12];
  float* out = (float*)d_out;

  int K = (int)(SS * 0.6224593312018546);   // 2549
  if (K < 1) K = 1; if (K > SS) K = SS;

  char* ws = (char*)d_ws;
  size_t off = 0;
  auto take = [&](size_t bytes) { void* p = ws + off; off += (bytes + 255) & ~size_t(255); return p; };
  __bf16* scaf = (__bf16*)take((size_t)BB * S2S * DD * 2);      // 16.8 MB
  __bf16* kb   = (__bf16*)take((size_t)BB * S2S * DD * 2);      // 16.8 MB
  __bf16* vT   = (__bf16*)take((size_t)BB * DD * S2S * 2);      // 16.8 MB
  __bf16* qb   = (__bf16*)take((size_t)BB * KPAD * DD * 2);     // 21.0 MB
  __bf16* ob   = (__bf16*)take((size_t)BB * KPAD * DD * 2);     // 21.0 MB
  float* scores = (float*)take((size_t)BB * SS * 4);
  float* gates  = (float*)take((size_t)BB * SS * 4);
  int*   idxb   = (int*)take((size_t)BB * KPAD * 4);
  float* conf   = (float*)take(256);

  k_init<<<1, 32, 0, stream>>>(conf);
  k_scaf_proj<<<dim3((BB * S2S) / 32, DD / 32), 32, 0, stream>>>(scafh, spw, spb, scaf);
  k_conf<<<BB * S2S, 256, 0, stream>>>(scaf, conf);
  k_scores<<<(BB * SS * 32) / 256, 256, 0, stream>>>(base, tw, tb, gw, gb, scores, gates);
  k_topk<<<BB, 1024, 0, stream>>>(scores, idxb, K);
  k_qproj<<<dim3(BB * QT32, DD / 32), 32, 0, stream>>>(base, idxb, ipw, ipb, qb);
  k_kvproj<<<dim3((BB * S2S) / 32, (2 * DD) / 32), 32, 0, stream>>>(
      scaf, ipw + (size_t)DD * DD, ipb + DD, kb, vT);
  k_attn<<<BB * HH * QT16, 32, 0, stream>>>(qb, kb, vT, ob);
  k_fill<<<(BB * SS * DD) / 256, 256, 0, stream>>>(base, gates, conf, thr, out);
  k_oproj_fuse<<<dim3(BB * QT32, DD / 32), 32, 0, stream>>>(
      ob, opw, opb, idxb, gates, base, conf, thr, out, K);
}
